// LatentMixtureAllRNNAgent_48954037240524
// MI455X (gfx1250) — matrix operand order repaired and checked by reading, not compiled, not run on hardware
//
#include <hip/hip_runtime.h>

typedef float  v8f   __attribute__((ext_vector_type(8)));
typedef __bf16 v16bf __attribute__((ext_vector_type(16)));

#define N_BATCH  4096
#define N_AG     8
#define DIN      256
#define DH       256
#define DACT     30
#define ROWS     64     // batch rows per block
#define LDX      264    // padded bf16 LDS row stride (bank-conflict free, 16B-aligned runs)
#define NTHREADS 256    // 8 wave32

// ---------------------------------------------------------------- WMMA helpers

static __device__ __forceinline__ v8f wmma_bf16(v16bf a, v16bf b, v8f c) {
  // D = A(16x32 bf16) * B(32x16 bf16) + C(16x16 f32)
  return __builtin_amdgcn_wmma_f32_16x16x32_bf16(false, a, false, b, (short)0, c,
                                                 false, false);
}

// A-fragment from LDS (row-major bf16, stride LDX). ISA layout:
// lane m=l&15 holds row m; sel=l>>4: K runs [kb+sel*8 .. +7] and [kb+16+sel*8 .. +7].
static __device__ __forceinline__ v16bf frag_a_lds(const __bf16* buf, int row,
                                                   int kb, int sel) {
  union { v16bf v; uint4 u[2]; } t;
  const __bf16* p = buf + row * LDX + kb + sel * 8;
  t.u[0] = *(const uint4*)(p);        // K = kb+sel*8 .. +7
  t.u[1] = *(const uint4*)(p + 16);   // K = kb+16+sel*8 .. +7
  return t.v;
}

// B-fragment from global bf16 weights stored transposed [n][k] (k contiguous).
// lane n=l&15 holds column n; sel=l>>4: K runs [kb+sel*8 .. +7] and [kb+16+sel*8 .. +7].
static __device__ __forceinline__ v16bf frag_b(const __bf16* wrow, int kb, int sel) {
  union { v16bf v; uint4 u[2]; } t;
  const __bf16* p = wrow + kb + sel * 8;
  t.u[0] = *(const uint4*)(p);
  t.u[1] = *(const uint4*)(p + 16);
  return t.v;
}

static __device__ __forceinline__ float sigmoid_f(float x) {
  return 1.0f / (1.0f + __expf(-x));
}
static __device__ __forceinline__ float tanh_f(float x) {
  float t = __expf(-2.0f * x);
  return (1.0f - t) / (1.0f + t);
}

// ------------------------------------------- prepass: transpose + cvt weights
// in: [A][K][N] f32  ->  out: [A][Npad][K] bf16 (rows n >= N zero-filled)
__global__ __launch_bounds__(256) void transpose_cvt_bf16(
    const float* __restrict__ in, __bf16* __restrict__ out,
    int K, int N, int Npad) {
  __shared__ float tile[32][33];
  const int a  = blockIdx.z;
  const int k0 = blockIdx.x * 32;
  const int n0 = blockIdx.y * 32;
  const int tx = threadIdx.x;      // 0..31
  const int ty = threadIdx.y;      // 0..7
  const float* inA = in + (long)a * K * N;
  __bf16* outA = out + (long)a * Npad * K;
#pragma unroll
  for (int i = 0; i < 32; i += 8) {
    int n = n0 + tx;
    float v = (n < N) ? inA[(long)(k0 + ty + i) * N + n] : 0.0f;
    tile[ty + i][tx] = v;
  }
  __syncthreads();
#pragma unroll
  for (int i = 0; i < 32; i += 8) {
    // out[n0+ty+i][k0+tx] = in[k0+tx][n0+ty+i]
    outA[(long)(n0 + ty + i) * K + (k0 + tx)] = (__bf16)tile[tx][ty + i];
  }
}

// ------------------------------------------------------------- fused main pass
__global__ __launch_bounds__(NTHREADS) void fused_gru_agent(
    const float* __restrict__ inp,     // [BS*A, 256]  row = b*A + a
    const float* __restrict__ h_in,    // [BS*A, 256]
    const float* __restrict__ fc1_b,   // [A,256]
    const float* __restrict__ ih_b,    // [A,768]
    const float* __restrict__ hh_b,    // [A,768]
    const float* __restrict__ fc2_b,   // [A,30]
    const __bf16* __restrict__ W1t,    // [A][256n][256k]
    const __bf16* __restrict__ Wiht,   // [A][768n][256k]
    const __bf16* __restrict__ Whht,   // [A][768n][256k]
    const __bf16* __restrict__ W2t,    // [A][32n][256k] (pad 30->32)
    float* __restrict__ out_q,         // [BS*A, 30]
    float* __restrict__ out_h) {       // [BS*A, 256]
  extern __shared__ __bf16 smem[];
  __bf16* inb_lds  = smem;                 // input tile bf16 (reused as h-out)
  __bf16* x_lds    = smem + ROWS * LDX;    // fc1 output bf16
  __bf16* hin_lds  = smem + 2 * ROWS * LDX;
  __bf16* hout_lds = inb_lds;              // reuse after phase 1

  const int a       = blockIdx.y;
  const int rowbase = blockIdx.x * ROWS;
  const int tid     = threadIdx.x;
  const int w       = tid >> 5;
  const int l       = tid & 31;
  const int ml      = l & 15;   // row (A) / col (B,C) within 16-tile
  const int sel     = l >> 4;

  const __bf16* W1a  = W1t  + (long)a * DH * DIN;
  const __bf16* Wiha = Wiht + (long)a * 3 * DH * DH;
  const __bf16* Whha = Whht + (long)a * 3 * DH * DH;
  const __bf16* W2a  = W2t  + (long)a * 32 * DH;

  // ---- phase 0: stage input + h_in tiles into LDS as bf16 (coalesced) ----
  for (int e = tid; e < ROWS * DH; e += NTHREADS) {
    int r = e >> 8, c = e & 255;
    long grow = (long)(rowbase + r) * N_AG + a;
    inb_lds[r * LDX + c] = (__bf16)inp[grow * DIN + c];
    hin_lds[r * LDX + c] = (__bf16)h_in[grow * DH + c];
  }
  __syncthreads();

  // ---- phase 1: x = relu(inp @ W1 + b1), kept bf16 in LDS ----
  for (int t = w; t < 64; t += 8) {          // 4 row-tiles x 16 col-tiles
    int rt = t >> 4, ct = t & 15;
    const __bf16* brow = W1a + (long)(ct * 16 + ml) * DIN;
    v8f acc = {};
#pragma unroll
    for (int kt = 0; kt < 8; ++kt) {
      v16bf av = frag_a_lds(inb_lds, rt * 16 + ml, kt * 32, sel);
      v16bf bv = frag_b(brow, kt * 32, sel);
      acc = wmma_bf16(av, bv, acc);
    }
    float bias = fc1_b[a * DH + ct * 16 + ml];
    __bf16* xo = x_lds + (rt * 16 + sel * 8) * LDX + ct * 16 + ml;
#pragma unroll
    for (int i = 0; i < 8; ++i) {
      float v = acc[i] + bias;
      xo[i * LDX] = (__bf16)(v > 0.0f ? v : 0.0f);
    }
  }
  __syncthreads();

  // ---- phase 2: six gate GEMM tiles + GRU combine, h -> global + LDS ----
  for (int t = w; t < 64; t += 8) {
    int rt = t >> 4, ct = t & 15;
    const int n = ct * 16 + ml;
    const __bf16* b_ir = Wiha + (long)(0 * DH + n) * DH;
    const __bf16* b_ii = Wiha + (long)(1 * DH + n) * DH;
    const __bf16* b_in = Wiha + (long)(2 * DH + n) * DH;
    const __bf16* b_hr = Whha + (long)(0 * DH + n) * DH;
    const __bf16* b_hi = Whha + (long)(1 * DH + n) * DH;
    const __bf16* b_hn = Whha + (long)(2 * DH + n) * DH;
    v8f Gir = {}, Gii = {}, Gin = {}, Ghr = {}, Ghi = {}, Ghn = {};
#pragma unroll
    for (int kt = 0; kt < 8; ++kt) {
      int kb = kt * 32;
      v16bf ax = frag_a_lds(x_lds,  rt * 16 + ml, kb, sel);
      v16bf ah = frag_a_lds(hin_lds, rt * 16 + ml, kb, sel);
      Gir = wmma_bf16(ax, frag_b(b_ir, kb, sel), Gir);
      Gii = wmma_bf16(ax, frag_b(b_ii, kb, sel), Gii);
      Gin = wmma_bf16(ax, frag_b(b_in, kb, sel), Gin);
      Ghr = wmma_bf16(ah, frag_b(b_hr, kb, sel), Ghr);
      Ghi = wmma_bf16(ah, frag_b(b_hi, kb, sel), Ghi);
      Ghn = wmma_bf16(ah, frag_b(b_hn, kb, sel), Ghn);
    }
    float bir = ih_b[a * 3 * DH + 0 * DH + n];
    float bii = ih_b[a * 3 * DH + 1 * DH + n];
    float bin = ih_b[a * 3 * DH + 2 * DH + n];
    float bhr = hh_b[a * 3 * DH + 0 * DH + n];
    float bhi = hh_b[a * 3 * DH + 1 * DH + n];
    float bhn = hh_b[a * 3 * DH + 2 * DH + n];
    __bf16* ho = hout_lds + (rt * 16 + sel * 8) * LDX + ct * 16 + ml;
#pragma unroll
    for (int i = 0; i < 8; ++i) {
      long grow = (long)(rowbase + rt * 16 + i + sel * 8) * N_AG + a;
      float hprev = h_in[grow * DH + n];                    // f32 for exactness
      float r  = sigmoid_f(Gir[i] + bir + Ghr[i] + bhr);
      float z  = sigmoid_f(Gii[i] + bii + Ghi[i] + bhi);
      float ng = tanh_f(Gin[i] + bin + r * (Ghn[i] + bhn));
      float h  = ng + z * (hprev - ng);
      out_h[grow * DH + n] = h;
      ho[i * LDX] = (__bf16)h;
    }
  }
  __syncthreads();

  // ---- phase 3: q = relu(h @ W2 + b2), N=30 padded to 32 ----
  {
    int rt = w >> 1, ct = w & 1;          // 4 row-tiles x 2 col-tiles, 1/wave
    const int n = ct * 16 + ml;           // 0..31
    const __bf16* brow = W2a + (long)n * DH;
    v8f acc = {};
#pragma unroll
    for (int kt = 0; kt < 8; ++kt) {
      v16bf av = frag_a_lds(hout_lds, rt * 16 + ml, kt * 32, sel);
      acc = wmma_bf16(av, frag_b(brow, kt * 32, sel), acc);
    }
    if (n < DACT) {
      float bias = fc2_b[a * DACT + n];
#pragma unroll
      for (int i = 0; i < 8; ++i) {
        long grow = (long)(rowbase + rt * 16 + i + sel * 8) * N_AG + a;
        float v = acc[i] + bias;
        out_q[grow * DACT + n] = v > 0.0f ? v : 0.0f;
      }
    }
  }
}

// ------------------------------------------------------------------ launcher
extern "C" void kernel_launch(void* const* d_in, const int* in_sizes, int n_in,
                              void* d_out, int out_size, void* d_ws, size_t ws_size,
                              hipStream_t stream) {
  (void)in_sizes; (void)n_in; (void)out_size; (void)ws_size;
  const float* inp   = (const float*)d_in[0];
  const float* hid   = (const float*)d_in[1];
  const float* fc1_w = (const float*)d_in[2];
  const float* fc1_b = (const float*)d_in[3];
  const float* ih_w  = (const float*)d_in[4];
  const float* ih_b  = (const float*)d_in[5];
  const float* hh_w  = (const float*)d_in[6];
  const float* hh_b  = (const float*)d_in[7];
  const float* fc2_w = (const float*)d_in[8];
  const float* fc2_b = (const float*)d_in[9];

  float* out_q = (float*)d_out;                          // [32768, 30]
  float* out_h = out_q + (size_t)N_BATCH * N_AG * DACT;  // [32768, 256]

  // workspace: bf16 transposed weights (≈7.1 MB)
  __bf16* ws   = (__bf16*)d_ws;
  __bf16* W1t  = ws;                       // 8*256*256
  __bf16* Wiht = W1t  + (size_t)8 * 256 * 256;   // 8*768*256
  __bf16* Whht = Wiht + (size_t)8 * 768 * 256;   // 8*768*256
  __bf16* W2t  = Whht + (size_t)8 * 768 * 256;   // 8*32*256

  dim3 tb(32, 8);
  transpose_cvt_bf16<<<dim3(8, 8, 8),  tb, 0, stream>>>(fc1_w, W1t, 256, 256, 256);
  transpose_cvt_bf16<<<dim3(8, 24, 8), tb, 0, stream>>>(ih_w, Wiht, 256, 768, 768);
  transpose_cvt_bf16<<<dim3(8, 24, 8), tb, 0, stream>>>(hh_w, Whht, 256, 768, 768);
  transpose_cvt_bf16<<<dim3(8, 1, 8),  tb, 0, stream>>>(fc2_w, W2t, 256, 30, 32);

  size_t shmem = (size_t)3 * ROWS * LDX * sizeof(__bf16);  // ~101 KB of 320 KB/WGP
  fused_gru_agent<<<dim3(N_BATCH / ROWS, N_AG), NTHREADS, shmem, stream>>>(
      inp, hid, fc1_b, ih_b, hh_b, fc2_b, W1t, Wiht, Whht, W2t, out_q, out_h);
}
